// Attention_2826088481351
// MI455X (gfx1250) — matrix operand order repaired and checked by reading, not compiled, not run on hardware
//
#include <hip/hip_runtime.h>
#include <math.h>

// ---------------------------------------------------------------------------
// Types for CDNA5 WMMA (wave32): bf16 A/B fragments, f32 accumulator
// ---------------------------------------------------------------------------
typedef __attribute__((ext_vector_type(16))) __bf16 v16bf;
typedef __attribute__((ext_vector_type(8)))  __bf16 v8bf;
typedef __attribute__((ext_vector_type(8)))  float  v8f;

// TDM descriptor SGPR groups (6-arg builtin: g0 u32x4, g1 i32x8, i32x4, i32x4, i32x8, cpol)
typedef __attribute__((ext_vector_type(4))) unsigned int u32x4;
typedef __attribute__((ext_vector_type(8))) int          i32x8;
typedef __attribute__((ext_vector_type(4))) int          i32x4;

#if defined(__HIP_DEVICE_COMPILE__) && __has_builtin(__builtin_amdgcn_tensor_load_to_lds)
#define USE_TDM 1
#else
#define USE_TDM 0
#endif

#define BATCH      2
#define SEQ        2048
#define DMODEL     4096
#define NHEADS     32
#define NKVHEADS   8
#define HEADDIM    128
#define KVDIM      (NKVHEADS * HEADDIM)   // 1024

// ---------------------------------------------------------------------------
// f32 -> bf16 cast
// ---------------------------------------------------------------------------
__global__ void cast_f32_bf16(const float* __restrict__ in, __bf16* __restrict__ out, int n) {
    int i = blockIdx.x * blockDim.x + threadIdx.x;
    if (i < n) out[i] = (__bf16)in[i];
}

// ---------------------------------------------------------------------------
// Tiled WMMA GEMM:  C[M,N] = A[M,K] * Bw[N,K]^T   (A,Bw bf16 row-major)
//
// Block = 8 waves (256 thr): block tile 128(M) x 64(N), k-step 32.
// The 64x32 bf16 B tile is shared by all 8 waves and staged in LDS by the
// Tensor Data Mover (tensor_load_to_lds, D# per CDNA5 ISA), double
// buffered: wave 0 issues the k+32 DMA, waits TENSORcnt<=1 (current tile
// done, in-order completion), then a workgroup barrier publishes the tile.
// A fragments stay direct-from-global (no cross-wave reuse available).
//
// Fragment layouts used here (contiguous k, same order for A and B):
//   A 16x32:  lanes 0-15 -> M rows;   lane half selects K {0..15} / {16..31}
//   B 32x16:  lane&15 -> N column;    lane half selects K {0..15} / {16..31}
//   C 16x16:  lane&15 -> N column;    VGPR r -> M = r + 8*(lane>=16)
// ---------------------------------------------------------------------------
template <typename CT>
__global__ __launch_bounds__(256)
void gemm_wmma_bf16(const __bf16* __restrict__ A,
                    const __bf16* __restrict__ Bw,
                    CT* __restrict__ C,
                    int M, int N, int K) {
    const int lane = threadIdx.x;            // 0..31
    const int wv   = threadIdx.y;            // 0..7
    const int m0   = blockIdx.y * 128 + wv * 16;
    const int n0   = blockIdx.x * 64;

    const int row  = lane & 15;
    const int half = lane >> 4;

    __shared__ __bf16 sB[2][64][32];         // [buf][n][k] : 2 x 4 KB

    v8f acc[4] = {};
    const __bf16* ap = A + (size_t)(m0 + row) * K + half * 16;

    // one k-step of compute: A frag from global, 4 B frags from LDS tile
    auto compute = [&](int buf, int k0) {
        v8bf lo = *(const v8bf*)(ap + k0);
        v8bf hi = *(const v8bf*)(ap + k0 + 8);
        v16bf a;
#pragma unroll
        for (int i = 0; i < 8; ++i) { a[i] = lo[i]; a[8 + i] = hi[i]; }
        __builtin_prefetch(ap + k0 + 128, 0, 3);      // global_prefetch_b8
#pragma unroll
        for (int t = 0; t < 4; ++t) {
            v16bf b = *(const v16bf*)&sB[buf][t * 16 + row][half * 16];
            acc[t] = __builtin_amdgcn_wmma_f32_16x16x32_bf16(
                false, a, false, b, (short)0, acc[t], false, false);
        }
    };

#if USE_TDM
    // D# group 1 (constant across k): data_size=2B, tensor [dim1=N, dim0=K],
    // tile [dim1=64, dim0=32], dim0 stride = K elements.
    i32x8 g1;
    g1[0] = 0x00010000;                                        // data_size code 1
    g1[1] = (int)((unsigned)(K & 0xFFFF) << 16);               // tensor_dim0[15:0]
    g1[2] = (int)(((unsigned)K >> 16) | ((unsigned)(N & 0xFFFF) << 16)); // dim0 hi | dim1 lo
    g1[3] = (int)(((unsigned)N >> 16) | (32u << 16));          // dim1 hi | tile_dim0=32
    g1[4] = 64;                                                // tile_dim1=64
    g1[5] = K;                                                 // tensor_dim0_stride[31:0]
    g1[6] = 0;
    g1[7] = 0;
    i32x4 gz4 = {};
    i32x8 gz8 = {};

    auto issue = [&](int kk, int buf) {
        unsigned long long ga =
            (unsigned long long)(size_t)(const void*)(Bw + (size_t)n0 * K + kk);
        unsigned ldsoff = (unsigned)(size_t)(void*)&sB[buf][0][0]; // low 32b of flat = LDS offset
        u32x4 g0;
        g0[0] = 1u;                                  // count=1, user mode
        g0[1] = ldsoff;                              // lds_addr
        g0[2] = (unsigned)ga;                        // global_addr[31:0]
        g0[3] = (unsigned)(ga >> 32) | (2u << 30);   // global_addr[56:32] | type=2
        __builtin_amdgcn_tensor_load_to_lds(g0, g1, gz4, gz4, gz8, 0);
    };

    if (wv == 0) issue(0, 0);
    int parity = 0;
    for (int k0 = 0; k0 < K; k0 += 32) {
        if (wv == 0) {
            if (k0 + 32 < K) {
                issue(k0 + 32, parity ^ 1);                 // prefetch next tile
                __builtin_amdgcn_s_wait_tensorcnt(1);       // current tile landed
            } else {
                __builtin_amdgcn_s_wait_tensorcnt(0);
            }
        }
        __syncthreads();                                    // publish buf[parity]
        compute(parity, k0);
        __syncthreads();                                    // done reading before refill
        parity ^= 1;
    }
#else
    // Cooperative-copy fallback (also what the host pass parses)
    const int tid = wv * 32 + lane;                         // 0..255
    for (int k0 = 0; k0 < K; k0 += 32) {
        int idx = tid * 8;                                  // 2048 elems / 256 thr
        int n = idx >> 5, c = idx & 31;
        *(v8bf*)&sB[0][n][c] = *(const v8bf*)(Bw + (size_t)(n0 + n) * K + k0 + c);
        __syncthreads();
        compute(0, k0);
        __syncthreads();
    }
#endif

    const int col   = lane & 15;
    const int rbase = half * 8;
#pragma unroll
    for (int t = 0; t < 4; ++t)
#pragma unroll
        for (int r = 0; r < 8; ++r)
            C[(size_t)(m0 + rbase + r) * N + n0 + t * 16 + col] = (CT)acc[t][r];
}

// ---------------------------------------------------------------------------
// RoPE + head transpose:  in [B,S,H,128] bf16 -> out [B,H,S,128] bf16
// ---------------------------------------------------------------------------
__global__ void rope_transpose(const __bf16* __restrict__ in, __bf16* __restrict__ out,
                               const float* __restrict__ cosT, const float* __restrict__ sinT,
                               int H, int total) {
    int idx = blockIdx.x * blockDim.x + threadIdx.x;
    if (idx >= total) return;
    int d2 = idx & 63;
    int h  = (idx >> 6) % H;
    int s  = (idx / (64 * H)) % SEQ;
    int b  = idx / (64 * H * SEQ);

    size_t pin  = (((size_t)(b * SEQ + s) * H + h) * HEADDIM) + 2 * d2;
    float te = (float)in[pin];
    float to = (float)in[pin + 1];
    float c  = cosT[s * 64 + d2];
    float si = sinT[s * 64 + d2];

    size_t pout = (((size_t)(b * H + h) * SEQ + s) * HEADDIM) + 2 * d2;
    out[pout]     = (__bf16)(te * c - to * si);
    out[pout + 1] = (__bf16)(te * si + to * c);
}

// V: [B,S,KVH,128] bf16 -> Vt [B,KVH,128,S] bf16
__global__ void transpose_v(const __bf16* __restrict__ in, __bf16* __restrict__ out, int total) {
    int idx = blockIdx.x * blockDim.x + threadIdx.x;
    if (idx >= total) return;
    int d = idx & 127;
    int h = (idx >> 7) % NKVHEADS;
    int s = (idx / (128 * NKVHEADS)) % SEQ;
    int b = idx / (128 * NKVHEADS * SEQ);
    size_t pin  = (((size_t)(b * SEQ + s) * NKVHEADS + h) * HEADDIM) + d;
    size_t pout = (((size_t)(b * NKVHEADS + h) * HEADDIM + d) * SEQ) + s;
    out[pout] = in[pin];
}

// ---------------------------------------------------------------------------
// Flash attention, causal, GQA (n_rep = 4). One wave per (b, h, 16-query tile).
// ---------------------------------------------------------------------------
__global__ __launch_bounds__(32)
void flash_attn(const __bf16* __restrict__ Qh,   // [B,H,S,128]
                const __bf16* __restrict__ Kh,   // [B,KVH,S,128]
                const __bf16* __restrict__ Vt,   // [B,KVH,128,S]
                __bf16* __restrict__ Ob) {       // [B,S,H*128]
    const int lane = threadIdx.x;
    const int qt   = blockIdx.x;
    const int h    = blockIdx.y;
    const int b    = blockIdx.z;
    const int kvh  = h >> 2;
    const int s0   = qt * 16;

    const int row   = lane & 15;
    const int half  = lane >> 4;
    const int rbase = half * 8;

    __shared__ float sP[16][32];

    const __bf16* qbase = Qh + (((size_t)(b * NHEADS + h) * SEQ) + s0) * HEADDIM;
    const __bf16* kbase = Kh + ((size_t)(b * NKVHEADS + kvh) * SEQ) * HEADDIM;
    const __bf16* vbase = Vt + ((size_t)(b * NKVHEADS + kvh) * HEADDIM) * SEQ;

    v16bf qf[4];
#pragma unroll
    for (int kc = 0; kc < 4; ++kc) {
        const __bf16* p = qbase + (size_t)row * HEADDIM + kc * 32 + half * 16;
        v8bf lo = *(const v8bf*)p;
        v8bf hi = *(const v8bf*)(p + 8);
#pragma unroll
        for (int i = 0; i < 8; ++i) { qf[kc][i] = lo[i]; qf[kc][8 + i] = hi[i]; }
    }

    v8f o[8] = {};
    float mrow[8], lrow[8];
#pragma unroll
    for (int r = 0; r < 8; ++r) { mrow[r] = -1e30f; lrow[r] = 0.0f; }

    const float scale = 0.08838834764831845f;   // 1/sqrt(128)

    for (int k0 = 0; k0 <= s0 + 15; k0 += 32) {
        v8f sc[2] = {};
#pragma unroll
        for (int ns = 0; ns < 2; ++ns) {
#pragma unroll
            for (int kc = 0; kc < 4; ++kc) {
                const __bf16* bp = kbase + (size_t)(k0 + ns * 16 + row) * HEADDIM
                                         + kc * 32 + half * 16;
                v16bf bf = *(const v16bf*)bp;
                sc[ns] = __builtin_amdgcn_wmma_f32_16x16x32_bf16(
                    false, qf[kc], false, bf, (short)0, sc[ns], false, false);
            }
        }

#pragma unroll
        for (int ns = 0; ns < 2; ++ns) {
            int kpos = k0 + ns * 16 + row;
#pragma unroll
            for (int r = 0; r < 8; ++r) {
                int qpos = s0 + rbase + r;
                float v = sc[ns][r] * scale;
                sc[ns][r] = (kpos > qpos) ? -1e30f : v;
            }
        }

#pragma unroll
        for (int r = 0; r < 8; ++r) {
            float mx = fmaxf(sc[0][r], sc[1][r]);
#pragma unroll
            for (int msk = 1; msk <= 8; msk <<= 1)
                mx = fmaxf(mx, __shfl_xor(mx, msk, 32));
            float mnew  = fmaxf(mrow[r], mx);
            float alpha = __expf(mrow[r] - mnew);
            float p0 = __expf(sc[0][r] - mnew);
            float p1 = __expf(sc[1][r] - mnew);
            float rs = p0 + p1;
#pragma unroll
            for (int msk = 1; msk <= 8; msk <<= 1)
                rs += __shfl_xor(rs, msk, 32);
            lrow[r] = lrow[r] * alpha + rs;
            mrow[r] = mnew;
            sc[0][r] = p0;
            sc[1][r] = p1;
#pragma unroll
            for (int t = 0; t < 8; ++t) o[t][r] *= alpha;
        }

        // C-layout -> A-layout transpose of P through LDS (single wave: in-order DS)
#pragma unroll
        for (int ns = 0; ns < 2; ++ns)
#pragma unroll
            for (int r = 0; r < 8; ++r)
                sP[rbase + r][ns * 16 + row] = sc[ns][r];

        v16bf pa;
#pragma unroll
        for (int i = 0; i < 8; ++i) {
            pa[i]     = (__bf16)sP[row][half * 16 + i];
            pa[8 + i] = (__bf16)sP[row][half * 16 + 8 + i];
        }

#pragma unroll
        for (int t = 0; t < 8; ++t) {
            const __bf16* vp = vbase + (size_t)(t * 16 + row) * SEQ + k0 + half * 16;
            v16bf bf = *(const v16bf*)vp;
            o[t] = __builtin_amdgcn_wmma_f32_16x16x32_bf16(
                false, pa, false, bf, (short)0, o[t], false, false);
        }
    }

#pragma unroll
    for (int t = 0; t < 8; ++t)
#pragma unroll
        for (int r = 0; r < 8; ++r) {
            float v = o[t][r] / lrow[r];
            size_t s = (size_t)(b * SEQ + s0 + rbase + r);
            Ob[s * (size_t)DMODEL + h * HEADDIM + t * 16 + row] = (__bf16)v;
        }
}

// ---------------------------------------------------------------------------
// Host launcher
// ---------------------------------------------------------------------------
extern "C" void kernel_launch(void* const* d_in, const int* in_sizes, int n_in,
                              void* d_out, int out_size, void* d_ws, size_t ws_size,
                              hipStream_t stream) {
    (void)in_sizes; (void)n_in; (void)out_size; (void)ws_size;

    const float* x    = (const float*)d_in[0];
    const float* wq   = (const float*)d_in[1];
    const float* wk   = (const float*)d_in[2];
    const float* wv   = (const float*)d_in[3];
    const float* wo   = (const float*)d_in[4];
    const float* cosT = (const float*)d_in[5];
    const float* sinT = (const float*)d_in[6];
    float* out = (float*)d_out;

    const int M   = BATCH * SEQ;                       // 4096
    const int nX  = M * DMODEL;
    const int nWq = DMODEL * DMODEL;
    const int nWk = KVDIM * DMODEL;
    const int nQ  = M * DMODEL;
    const int nK  = M * KVDIM;

    char* p = (char*)d_ws;
    __bf16* xb  = (__bf16*)p;  p += (size_t)nX  * 2;
    __bf16* wqb = (__bf16*)p;  p += (size_t)nWq * 2;
    __bf16* wkb = (__bf16*)p;  p += (size_t)nWk * 2;
    __bf16* wvb = (__bf16*)p;  p += (size_t)nWk * 2;
    __bf16* wob = (__bf16*)p;  p += (size_t)nWq * 2;
    __bf16* Qb  = (__bf16*)p;  p += (size_t)nQ  * 2;  // [B,S,H,128]
    __bf16* Kb  = (__bf16*)p;  p += (size_t)nK  * 2;  // [B,S,KVH,128]
    __bf16* Vb  = (__bf16*)p;  p += (size_t)nK  * 2;
    __bf16* Qh  = (__bf16*)p;  p += (size_t)nQ  * 2;  // [B,H,S,128]
    __bf16* Kh  = (__bf16*)p;  p += (size_t)nK  * 2;  // [B,KVH,S,128]
    __bf16* Vt  = (__bf16*)p;  p += (size_t)nK  * 2;  // [B,KVH,128,S]
    __bf16* Ab  = xb;                                  // reuse xb for attn output

    const int T = 256;
    cast_f32_bf16<<<(nX  + T - 1) / T, T, 0, stream>>>(x,  xb,  nX);
    cast_f32_bf16<<<(nWq + T - 1) / T, T, 0, stream>>>(wq, wqb, nWq);
    cast_f32_bf16<<<(nWk + T - 1) / T, T, 0, stream>>>(wk, wkb, nWk);
    cast_f32_bf16<<<(nWk + T - 1) / T, T, 0, stream>>>(wv, wvb, nWk);
    cast_f32_bf16<<<(nWq + T - 1) / T, T, 0, stream>>>(wo, wob, nWq);

    dim3 gblk(32, 8);
    gemm_wmma_bf16<__bf16><<<dim3(DMODEL / 64, M / 128), gblk, 0, stream>>>(
        xb, wqb, Qb, M, DMODEL, DMODEL);
    gemm_wmma_bf16<__bf16><<<dim3(KVDIM / 64, M / 128), gblk, 0, stream>>>(
        xb, wkb, Kb, M, KVDIM, DMODEL);
    gemm_wmma_bf16<__bf16><<<dim3(KVDIM / 64, M / 128), gblk, 0, stream>>>(
        xb, wvb, Vb, M, KVDIM, DMODEL);

    int nRq = BATCH * SEQ * NHEADS * 64;
    int nRk = BATCH * SEQ * NKVHEADS * 64;
    rope_transpose<<<(nRq + T - 1) / T, T, 0, stream>>>(Qb, Qh, cosT, sinT, NHEADS, nRq);
    rope_transpose<<<(nRk + T - 1) / T, T, 0, stream>>>(Kb, Kh, cosT, sinT, NKVHEADS, nRk);
    int nTv = BATCH * SEQ * NKVHEADS * HEADDIM;
    transpose_v<<<(nTv + T - 1) / T, T, 0, stream>>>(Vb, Vt, nTv);

    flash_attn<<<dim3(SEQ / 16, NHEADS, BATCH), 32, 0, stream>>>(Qh, Kh, Vt, Ab);

    gemm_wmma_bf16<float><<<dim3(DMODEL / 64, M / 128), gblk, 0, stream>>>(
        Ab, wob, out, M, DMODEL, DMODEL);
}
